// DatabricksBlock_89833535963322
// MI455X (gfx1250) — compile-verified, matrix operand order checked by reading, not run on hardware
//
#include <hip/hip_runtime.h>
#include <hip/hip_bf16.h>

// ---------------------------------------------------------------------------
// Transformer block (attention + top-2 MoE) for MI455X / gfx1250.
// All matmuls go through v_wmma_f32_16x16x32_bf16 (wave32 WMMA).
// B=1, S=2048, D=2048, H=16, HKV=4, HD=128, E=8, TOPK=2, F=2048.
// ---------------------------------------------------------------------------

typedef __bf16 bf16;
typedef __attribute__((ext_vector_type(16))) __bf16 v16bf;
typedef __attribute__((ext_vector_type(8)))  float  v8f;

union AB { v16bf v; uint4 q[2]; };

static __device__ __forceinline__ v8f wmma_bf16(v16bf a, v16bf b, v8f c) {
  return __builtin_amdgcn_wmma_f32_16x16x32_bf16(false, a, false, b, (short)0, c,
                                                 false, false);
}
static __device__ __forceinline__ v8f vzero() {
  v8f v;
#pragma unroll
  for (int i = 0; i < 8; ++i) v[i] = 0.0f;
  return v;
}

constexpr int S_  = 2048;
constexpr int D_  = 2048;
constexpr int F_  = 2048;
constexpr int E_  = 8;
constexpr int H_  = 16;
constexpr int HKV_= 4;
constexpr int HD_ = 128;
constexpr int NQKV_ = (H_ + 2 * HKV_) * HD_;   // 3072

// ---------------------------------------------------------------------------
// float32 [R][C] -> bf16 [C][R] (transpose + convert), so every GEMM operand
// B is stored N-major ("BT") and tiles load identically to A tiles.
// ---------------------------------------------------------------------------
__global__ __launch_bounds__(256) void k_cvtT(const float* __restrict__ in,
                                              bf16* __restrict__ out,
                                              int R, int C) {
  int idx = blockIdx.x * 256 + threadIdx.x;
  int n = R * C;
  if (idx < n) {
    int r = idx / C, c = idx - r * C;
    out[(size_t)c * R + r] = (bf16)in[idx];
  }
}

// ---------------------------------------------------------------------------
// LayerNorm (optionally fused residual add): s = a (+ b); res_out = s;
// out = LN(s) in bf16. One token per block, 256 threads, D=2048.
// ---------------------------------------------------------------------------
__global__ __launch_bounds__(256) void k_addln(const float* __restrict__ a,
                                               const float* __restrict__ b,
                                               float* __restrict__ res_out,
                                               const float* __restrict__ w,
                                               const float* __restrict__ bias,
                                               bf16* __restrict__ out) {
  __shared__ float red[256];
  const int s = blockIdx.x, tid = threadIdx.x;
  const size_t base = (size_t)s * D_;
  float vals[8];
  float sum = 0.f;
#pragma unroll
  for (int i = 0; i < 8; ++i) {
    int d = tid + i * 256;
    float v = a[base + d];
    if (b) v += b[base + d];
    vals[i] = v;
    sum += v;
    if (res_out) res_out[base + d] = v;
  }
  red[tid] = sum; __syncthreads();
  for (int st = 128; st > 0; st >>= 1) {
    if (tid < st) red[tid] += red[tid + st];
    __syncthreads();
  }
  float mu = red[0] / (float)D_;
  __syncthreads();
  float vs = 0.f;
#pragma unroll
  for (int i = 0; i < 8; ++i) { float d = vals[i] - mu; vs += d * d; }
  red[tid] = vs; __syncthreads();
  for (int st = 128; st > 0; st >>= 1) {
    if (tid < st) red[tid] += red[tid + st];
    __syncthreads();
  }
  float rstd = rsqrtf(red[0] / (float)D_ + 1e-5f);
#pragma unroll
  for (int i = 0; i < 8; ++i) {
    int d = tid + i * 256;
    out[base + d] = (bf16)((vals[i] - mu) * rstd * w[d] + bias[d]);
  }
}

// ---------------------------------------------------------------------------
// Generic bf16 WMMA GEMM: C[M][N] = A[M][K] * BT[N][K]^T, optional clip.
// Block = 8 waves, tile 128x128, wave tile 32x64 (2x4 WMMA accums), K step 32.
// ---------------------------------------------------------------------------
__global__ __launch_bounds__(256) void k_gemm(const bf16* __restrict__ A,
                                              const bf16* __restrict__ BT,
                                              float* __restrict__ C,
                                              int M, int N, int Kd, float clip) {
  __shared__ __align__(16) bf16 As[128][32];
  __shared__ __align__(16) bf16 Bs[128][32];
  const int m0 = blockIdx.x * 128, n0 = blockIdx.y * 128;
  const int tid = threadIdx.x, wave = tid >> 5, lane = tid & 31;
  const int l16 = lane & 15; const bool hi = lane >= 16;
  const int wm = wave & 3, wn = wave >> 2;

  v8f acc[2][4];
#pragma unroll
  for (int i = 0; i < 2; ++i)
#pragma unroll
    for (int j = 0; j < 4; ++j) acc[i][j] = vzero();

  for (int k0 = 0; k0 < Kd; k0 += 32) {
    __syncthreads();
#pragma unroll
    for (int i = 0; i < 2; ++i) {
      int idx = tid + i * 256;          // 512 x 16B covers both 128x32 tiles
      int r = idx >> 2, cq = idx & 3;
      *(uint4*)&As[r][cq * 8] =
          *(const uint4*)(A + (size_t)(m0 + r) * Kd + k0 + cq * 8);
      *(uint4*)&Bs[r][cq * 8] =
          *(const uint4*)(BT + (size_t)(n0 + r) * Kd + k0 + cq * 8);
    }
    __syncthreads();
    if (k0 + 32 < Kd) {                 // global_prefetch_b8 for the next tiles
      __builtin_prefetch(A + (size_t)(m0 + (tid >> 1)) * Kd + k0 + 32);
      __builtin_prefetch(BT + (size_t)(n0 + (tid >> 1)) * Kd + k0 + 32);
    }
    AB a[2], bfr[4];
    const int koff = hi ? 8 : 0;
#pragma unroll
    for (int mt = 0; mt < 2; ++mt) {
      int row = wm * 32 + mt * 16 + l16;
      a[mt].q[0] = *(const uint4*)&As[row][koff];
      a[mt].q[1] = *(const uint4*)&As[row][koff + 16];
    }
#pragma unroll
    for (int nt = 0; nt < 4; ++nt) {
      int row = wn * 64 + nt * 16 + l16;
      bfr[nt].q[0] = *(const uint4*)&Bs[row][koff];
      bfr[nt].q[1] = *(const uint4*)&Bs[row][koff + 16];
    }
#pragma unroll
    for (int mt = 0; mt < 2; ++mt)
#pragma unroll
      for (int nt = 0; nt < 4; ++nt)
        acc[mt][nt] = wmma_bf16(a[mt].v, bfr[nt].v, acc[mt][nt]);
  }

#pragma unroll
  for (int mt = 0; mt < 2; ++mt)
#pragma unroll
    for (int nt = 0; nt < 4; ++nt)
#pragma unroll
      for (int r = 0; r < 8; ++r) {
        int row = m0 + wm * 32 + mt * 16 + (hi ? r + 8 : r);
        int col = n0 + wn * 64 + nt * 16 + l16;
        float v = acc[mt][nt][r];
        if (clip > 0.f) v = fminf(fmaxf(v, -clip), clip);
        C[(size_t)row * N + col] = v;
      }
}

// ---------------------------------------------------------------------------
// RoPE + split qkv(f32, clipped) into q/k (rotated) and v, all bf16.
// grid (S, 24), block 64. hh<16: q-head; 16..19: k-head; 20..23: v-head.
// ---------------------------------------------------------------------------
__global__ __launch_bounds__(64) void k_rope(const float* __restrict__ qkv,
                                             const int* __restrict__ pos,
                                             bf16* __restrict__ Qo,
                                             bf16* __restrict__ Ko,
                                             bf16* __restrict__ Vo) {
  const int s = blockIdx.x, hh = blockIdx.y, j = threadIdx.x;  // j < 64
  const float* row = qkv + (size_t)s * NQKV_;
  if (hh < H_) {
    float x1 = row[hh * HD_ + j], x2 = row[hh * HD_ + 64 + j];
    float fr = __expf(-((float)(2 * j) / (float)HD_) * __logf(500000.0f));
    float ang = (float)pos[s] * fr;
    float c = __cosf(ang), sn = __sinf(ang);
    bf16* o = Qo + (size_t)s * (H_ * HD_) + hh * HD_;
    o[j] = (bf16)(x1 * c - x2 * sn);
    o[j + 64] = (bf16)(x2 * c + x1 * sn);
  } else if (hh < H_ + HKV_) {
    int kvh = hh - H_;
    int base = H_ * HD_ + kvh * HD_;
    float x1 = row[base + j], x2 = row[base + 64 + j];
    float fr = __expf(-((float)(2 * j) / (float)HD_) * __logf(500000.0f));
    float ang = (float)pos[s] * fr;
    float c = __cosf(ang), sn = __sinf(ang);
    bf16* o = Ko + (size_t)s * (HKV_ * HD_) + kvh * HD_;
    o[j] = (bf16)(x1 * c - x2 * sn);
    o[j + 64] = (bf16)(x2 * c + x1 * sn);
  } else {
    int kvh = hh - H_ - HKV_;
    const float* vsrc = row + (H_ + HKV_) * HD_ + kvh * HD_;
    bf16* o = Vo + (size_t)s * (HKV_ * HD_) + kvh * HD_;
    o[j] = (bf16)vsrc[j];
    o[j + 64] = (bf16)vsrc[j + 64];
  }
}

// ---------------------------------------------------------------------------
// Flash attention, causal, GQA (H=16, HKV=4, HD=128).
// Block = 8 waves, Q-block 128 (16 rows/wave), K-block 64.
// QK^T and PV via WMMA; P re-shaped C-layout -> A-layout through LDS.
// ---------------------------------------------------------------------------
__global__ __launch_bounds__(256) void k_fattn(const bf16* __restrict__ Q,
                                               const bf16* __restrict__ K,
                                               const bf16* __restrict__ V,
                                               bf16* __restrict__ O) {
  __shared__ __align__(16) bf16 Kt[64][HD_];   // [krow][hd]   16KB
  __shared__ __align__(16) bf16 Vt[HD_][64];   // [hd][krow]   16KB (transposed)
  __shared__ __align__(16) bf16 Pl[128][64];   // probs        16KB
  const int qb = blockIdx.x, h = blockIdx.y, kvh = h >> 2;
  const int tid = threadIdx.x, wave = tid >> 5, lane = tid & 31;
  const int l16 = lane & 15; const bool hi = lane >= 16;
  const int q0 = qb * 128 + wave * 16;
  const int koff = hi ? 8 : 0;

  // Q fragments stay in registers for the whole kernel (4 x 16x32 over HD).
  AB qf[4];
  const bf16* qrow = Q + (size_t)(q0 + l16) * (H_ * HD_) + h * HD_;
#pragma unroll
  for (int kk = 0; kk < 4; ++kk) {
    qf[kk].q[0] = *(const uint4*)(qrow + kk * 32 + koff);
    qf[kk].q[1] = *(const uint4*)(qrow + kk * 32 + koff + 16);
  }

  v8f o[8];
#pragma unroll
  for (int i = 0; i < 8; ++i) o[i] = vzero();
  float mprev[8], lsum[8];
#pragma unroll
  for (int r = 0; r < 8; ++r) { mprev[r] = -1e30f; lsum[r] = 0.f; }

  const int nkb = qb * 2 + 2;
  for (int kb = 0; kb < nkb; ++kb) {
    __syncthreads();
    // K tile (row-major) and V tile (transposed) into LDS.
#pragma unroll
    for (int i = 0; i < 4; ++i) {
      int idx = tid + i * 256;               // 1024 x 16B
      int r = idx >> 4, cq = idx & 15;
      *(uint4*)&Kt[r][cq * 8] =
          *(const uint4*)(K + (size_t)(kb * 64 + r) * (HKV_ * HD_) + kvh * HD_ + cq * 8);
      uint4 tmp =
          *(const uint4*)(V + (size_t)(kb * 64 + r) * (HKV_ * HD_) + kvh * HD_ + cq * 8);
      const bf16* t = (const bf16*)&tmp;
#pragma unroll
      for (int j = 0; j < 8; ++j) Vt[cq * 8 + j][r] = t[j];
    }
    __syncthreads();

    // scores S = Q K^T  (16x64 per wave, 16 WMMAs)
    v8f sx[4];
#pragma unroll
    for (int nt = 0; nt < 4; ++nt) sx[nt] = vzero();
#pragma unroll
    for (int nt = 0; nt < 4; ++nt) {
      int n = nt * 16 + l16;
#pragma unroll
      for (int kk = 0; kk < 4; ++kk) {
        AB bfr;
        bfr.q[0] = *(const uint4*)&Kt[n][kk * 32 + koff];
        bfr.q[1] = *(const uint4*)&Kt[n][kk * 32 + koff + 16];
        sx[nt] = wmma_bf16(qf[kk].v, bfr.v, sx[nt]);
      }
    }

    // online softmax (rows 0-7 in lanes 0-15, rows 8-15 in lanes 16-31)
    const float sc = 0.08838834764831845f;    // 1/sqrt(128)
    float alpha[8];
#pragma unroll
    for (int r = 0; r < 8; ++r) {
      int row = q0 + (hi ? r + 8 : r);
      float m = mprev[r];
#pragma unroll
      for (int nt = 0; nt < 4; ++nt) {
        int col = kb * 64 + nt * 16 + l16;
        float x = sx[nt][r] * sc;
        if (col > row) x = -1e30f;
        sx[nt][r] = x;
        m = fmaxf(m, x);
      }
#pragma unroll
      for (int off = 1; off < 16; off <<= 1) m = fmaxf(m, __shfl_xor(m, off, 32));
      alpha[r] = __expf(mprev[r] - m);
      float p = 0.f;
#pragma unroll
      for (int nt = 0; nt < 4; ++nt) {
        float e = __expf(sx[nt][r] - m);
        sx[nt][r] = e;
        p += e;
      }
#pragma unroll
      for (int off = 1; off < 16; off <<= 1) p += __shfl_xor(p, off, 32);
      lsum[r] = lsum[r] * alpha[r] + p;
      mprev[r] = m;
    }
#pragma unroll
    for (int nt2 = 0; nt2 < 8; ++nt2)
#pragma unroll
      for (int r = 0; r < 8; ++r) o[nt2][r] *= alpha[r];

    // P (C-layout) -> LDS -> A-layout fragments. Per-wave private region,
    // and DS ops are in-order within a wave, so no workgroup barrier needed.
#pragma unroll
    for (int nt = 0; nt < 4; ++nt)
#pragma unroll
      for (int r = 0; r < 8; ++r)
        Pl[wave * 16 + (hi ? r + 8 : r)][nt * 16 + l16] = (bf16)sx[nt][r];

    // O += P V  (16x128 per wave, 16 WMMAs)
#pragma unroll
    for (int kk2 = 0; kk2 < 2; ++kk2) {
      AB pf;
      int prow = wave * 16 + l16;
      pf.q[0] = *(const uint4*)&Pl[prow][kk2 * 32 + koff];
      pf.q[1] = *(const uint4*)&Pl[prow][kk2 * 32 + koff + 16];
#pragma unroll
      for (int nt2 = 0; nt2 < 8; ++nt2) {
        AB vf;
        int n = nt2 * 16 + l16;
        vf.q[0] = *(const uint4*)&Vt[n][kk2 * 32 + koff];
        vf.q[1] = *(const uint4*)&Vt[n][kk2 * 32 + koff + 16];
        o[nt2] = wmma_bf16(pf.v, vf.v, o[nt2]);
      }
    }
  }

#pragma unroll
  for (int r = 0; r < 8; ++r) {
    float inv = 1.0f / lsum[r];
    int row = q0 + (hi ? r + 8 : r);
#pragma unroll
    for (int nt2 = 0; nt2 < 8; ++nt2)
      O[(size_t)row * (H_ * HD_) + h * HD_ + nt2 * 16 + l16] =
          (bf16)(o[nt2][r] * inv);
  }
}

// ---------------------------------------------------------------------------
// Router: logits = x @ Wrouter (D x 8), softmax, top-2, normalized weights.
// ---------------------------------------------------------------------------
__global__ __launch_bounds__(256) void k_router(const bf16* __restrict__ X,
                                                const float* __restrict__ Wr,
                                                int* __restrict__ tokE,
                                                float* __restrict__ tokW) {
  __shared__ float red[E_][256];
  const int s = blockIdx.x, tid = threadIdx.x;
  float p[E_];
#pragma unroll
  for (int e = 0; e < E_; ++e) p[e] = 0.f;
  for (int d = tid; d < D_; d += 256) {
    float xv = (float)X[(size_t)s * D_ + d];
#pragma unroll
    for (int e = 0; e < E_; ++e) p[e] += xv * Wr[(size_t)d * E_ + e];
  }
#pragma unroll
  for (int e = 0; e < E_; ++e) red[e][tid] = p[e];
  __syncthreads();
  for (int st = 128; st > 0; st >>= 1) {
    if (tid < st)
#pragma unroll
      for (int e = 0; e < E_; ++e) red[e][tid] += red[e][tid + st];
    __syncthreads();
  }
  if (tid == 0) {
    float lg[E_], mx = -1e30f;
#pragma unroll
    for (int e = 0; e < E_; ++e) { lg[e] = red[e][0]; mx = fmaxf(mx, lg[e]); }
    float sum = 0.f;
#pragma unroll
    for (int e = 0; e < E_; ++e) { lg[e] = __expf(lg[e] - mx); sum += lg[e]; }
#pragma unroll
    for (int e = 0; e < E_; ++e) lg[e] /= sum;
    int i0 = 0;
#pragma unroll
    for (int e = 1; e < E_; ++e) if (lg[e] > lg[i0]) i0 = e;
    int i1 = (i0 == 0) ? 1 : 0;
#pragma unroll
    for (int e = 0; e < E_; ++e)
      if (e != i0 && lg[e] > lg[i1]) i1 = e;
    float nrm = lg[i0] + lg[i1];
    tokE[2 * s] = i0; tokE[2 * s + 1] = i1;
    tokW[2 * s] = lg[i0] / nrm; tokW[2 * s + 1] = lg[i1] / nrm;
  }
}

__global__ __launch_bounds__(256) void k_count(const int* __restrict__ tokE,
                                               int* __restrict__ counts) {
  int s = blockIdx.x * 256 + threadIdx.x;
  if (s < S_) {
    atomicAdd(&counts[tokE[2 * s]], 1);
    atomicAdd(&counts[tokE[2 * s + 1]], 1);
  }
}

__global__ void k_offsets(const int* __restrict__ counts,
                          int* __restrict__ offs, int* __restrict__ cursor) {
  int acc = 0;
  for (int e = 0; e < E_; ++e) { offs[e] = acc; cursor[e] = acc; acc += counts[e]; }
  offs[E_] = acc;
}

__global__ __launch_bounds__(256) void k_scatter(const int* __restrict__ tokE,
                                                 const float* __restrict__ tokW,
                                                 int* __restrict__ cursor,
                                                 int* __restrict__ perm,
                                                 float* __restrict__ pairw,
                                                 int* __restrict__ pairpos) {
  int s = blockIdx.x * 256 + threadIdx.x;
  if (s < S_) {
#pragma unroll
    for (int k = 0; k < 2; ++k) {
      int e = tokE[2 * s + k];
      int pos = atomicAdd(&cursor[e], 1);
      perm[pos] = s;
      pairw[pos] = tokW[2 * s + k];
      pairpos[2 * s + k] = pos;
    }
  }
}

// ---------------------------------------------------------------------------
// MoE up (gathered grouped GEMM): inter = silu(x@w1[e]) * (x@v1[e]), bf16.
// ---------------------------------------------------------------------------
__global__ __launch_bounds__(256) void k_moe_up(const bf16* __restrict__ X,
                                                const bf16* __restrict__ W1T,
                                                const bf16* __restrict__ V1T,
                                                const int* __restrict__ perm,
                                                const int* __restrict__ offs,
                                                bf16* __restrict__ inter) {
  __shared__ __align__(16) bf16 As[128][32];
  __shared__ __align__(16) bf16 B1s[128][32];
  __shared__ __align__(16) bf16 B2s[128][32];
  __shared__ int rowTok[128];
  const int e = blockIdx.z;
  const int off = offs[e], cnt = offs[e + 1] - offs[e];
  const int m0 = blockIdx.x * 128;
  if (m0 >= cnt) return;
  const int n0 = blockIdx.y * 128;
  const bf16* w1 = W1T + (size_t)e * F_ * D_;   // [F][D]
  const bf16* v1 = V1T + (size_t)e * F_ * D_;
  const int tid = threadIdx.x, wave = tid >> 5, lane = tid & 31;
  const int l16 = lane & 15; const bool hi = lane >= 16;
  const int wm = wave & 3, wn = wave >> 2;
  const int koff = hi ? 8 : 0;

  if (tid < 128) {
    int rr = m0 + tid; if (rr >= cnt) rr = cnt - 1;
    rowTok[tid] = perm[off + rr];
  }

  v8f a1[2][4], a2[2][4];
#pragma unroll
  for (int i = 0; i < 2; ++i)
#pragma unroll
    for (int j = 0; j < 4; ++j) { a1[i][j] = vzero(); a2[i][j] = vzero(); }

  for (int k0 = 0; k0 < D_; k0 += 32) {
    __syncthreads();
#pragma unroll
    for (int i = 0; i < 2; ++i) {
      int idx = tid + i * 256;
      int r = idx >> 2, cq = idx & 3;
      *(uint4*)&As[r][cq * 8] =
          *(const uint4*)(X + (size_t)rowTok[r] * D_ + k0 + cq * 8);
      *(uint4*)&B1s[r][cq * 8] =
          *(const uint4*)(w1 + (size_t)(n0 + r) * D_ + k0 + cq * 8);
      *(uint4*)&B2s[r][cq * 8] =
          *(const uint4*)(v1 + (size_t)(n0 + r) * D_ + k0 + cq * 8);
    }
    __syncthreads();
    AB af[2], b1f[4], b2f[4];
#pragma unroll
    for (int mt = 0; mt < 2; ++mt) {
      int row = wm * 32 + mt * 16 + l16;
      af[mt].q[0] = *(const uint4*)&As[row][koff];
      af[mt].q[1] = *(const uint4*)&As[row][koff + 16];
    }
#pragma unroll
    for (int nt = 0; nt < 4; ++nt) {
      int row = wn * 64 + nt * 16 + l16;
      b1f[nt].q[0] = *(const uint4*)&B1s[row][koff];
      b1f[nt].q[1] = *(const uint4*)&B1s[row][koff + 16];
      b2f[nt].q[0] = *(const uint4*)&B2s[row][koff];
      b2f[nt].q[1] = *(const uint4*)&B2s[row][koff + 16];
    }
#pragma unroll
    for (int mt = 0; mt < 2; ++mt)
#pragma unroll
      for (int nt = 0; nt < 4; ++nt) {
        a1[mt][nt] = wmma_bf16(af[mt].v, b1f[nt].v, a1[mt][nt]);
        a2[mt][nt] = wmma_bf16(af[mt].v, b2f[nt].v, a2[mt][nt]);
      }
  }

#pragma unroll
  for (int mt = 0; mt < 2; ++mt)
#pragma unroll
    for (int nt = 0; nt < 4; ++nt)
#pragma unroll
      for (int r = 0; r < 8; ++r) {
        int grow = m0 + wm * 32 + mt * 16 + (hi ? r + 8 : r);
        if (grow < cnt) {
          float z1 = a1[mt][nt][r], z2 = a2[mt][nt][r];
          float sig = 1.0f / (1.0f + __expf(-z1));
          int col = n0 + wn * 64 + nt * 16 + l16;
          inter[(size_t)(off + grow) * F_ + col] = (bf16)(z1 * sig * z2);
        }
      }
}

// ---------------------------------------------------------------------------
// MoE down: pout[pair] = (inter[pair] @ w2[e]) * pairw[pair]  (f32).
// ---------------------------------------------------------------------------
__global__ __launch_bounds__(256) void k_moe_down(const bf16* __restrict__ inter,
                                                  const bf16* __restrict__ W2T,
                                                  const float* __restrict__ pairw,
                                                  const int* __restrict__ offs,
                                                  float* __restrict__ pout) {
  __shared__ __align__(16) bf16 As[128][32];
  __shared__ __align__(16) bf16 Bs[128][32];
  const int e = blockIdx.z;
  const int off = offs[e], cnt = offs[e + 1] - offs[e];
  const int m0 = blockIdx.x * 128;
  if (m0 >= cnt) return;
  const int n0 = blockIdx.y * 128;
  const bf16* w2 = W2T + (size_t)e * D_ * F_;   // [D][F]
  const int tid = threadIdx.x, wave = tid >> 5, lane = tid & 31;
  const int l16 = lane & 15; const bool hi = lane >= 16;
  const int wm = wave & 3, wn = wave >> 2;
  const int koff = hi ? 8 : 0;

  v8f acc[2][4];
#pragma unroll
  for (int i = 0; i < 2; ++i)
#pragma unroll
    for (int j = 0; j < 4; ++j) acc[i][j] = vzero();

  for (int k0 = 0; k0 < F_; k0 += 32) {
    __syncthreads();
#pragma unroll
    for (int i = 0; i < 2; ++i) {
      int idx = tid + i * 256;
      int r = idx >> 2, cq = idx & 3;
      int rr = m0 + r; if (rr >= cnt) rr = cnt - 1;
      *(uint4*)&As[r][cq * 8] =
          *(const uint4*)(inter + (size_t)(off + rr) * F_ + k0 + cq * 8);
      *(uint4*)&Bs[r][cq * 8] =
          *(const uint4*)(w2 + (size_t)(n0 + r) * F_ + k0 + cq * 8);
    }
    __syncthreads();
    AB af[2], bf[4];
#pragma unroll
    for (int mt = 0; mt < 2; ++mt) {
      int row = wm * 32 + mt * 16 + l16;
      af[mt].q[0] = *(const uint4*)&As[row][koff];
      af[mt].q[1] = *(const uint4*)&As[row][koff + 16];
    }
#pragma unroll
    for (int nt = 0; nt < 4; ++nt) {
      int row = wn * 64 + nt * 16 + l16;
      bf[nt].q[0] = *(const uint4*)&Bs[row][koff];
      bf[nt].q[1] = *(const uint4*)&Bs[row][koff + 16];
    }
#pragma unroll
    for (int mt = 0; mt < 2; ++mt)
#pragma unroll
      for (int nt = 0; nt < 4; ++nt)
        acc[mt][nt] = wmma_bf16(af[mt].v, bf[nt].v, acc[mt][nt]);
  }

#pragma unroll
  for (int mt = 0; mt < 2; ++mt)
#pragma unroll
    for (int nt = 0; nt < 4; ++nt)
#pragma unroll
      for (int r = 0; r < 8; ++r) {
        int grow = m0 + wm * 32 + mt * 16 + (hi ? r + 8 : r);
        if (grow < cnt) {
          int col = n0 + wn * 64 + nt * 16 + l16;
          pout[(size_t)(off + grow) * D_ + col] =
              acc[mt][nt][r] * pairw[off + grow];
        }
      }
}

// ---------------------------------------------------------------------------
// out = residual(h2) + pair0 + pair1
// ---------------------------------------------------------------------------
__global__ __launch_bounds__(256) void k_combine(const float* __restrict__ h2,
                                                 const float* __restrict__ pout,
                                                 const int* __restrict__ pairpos,
                                                 float* __restrict__ out) {
  const int s = blockIdx.x, tid = threadIdx.x;
  const int p0 = pairpos[2 * s], p1 = pairpos[2 * s + 1];
#pragma unroll
  for (int i = 0; i < 8; ++i) {
    int d = tid + i * 256;
    out[(size_t)s * D_ + d] = h2[(size_t)s * D_ + d] +
                              pout[(size_t)p0 * D_ + d] +
                              pout[(size_t)p1 * D_ + d];
  }
}

// ---------------------------------------------------------------------------
// Host driver
// ---------------------------------------------------------------------------
extern "C" void kernel_launch(void* const* d_in, const int* in_sizes, int n_in,
                              void* d_out, int out_size, void* d_ws, size_t ws_size,
                              hipStream_t stream) {
  const float* hidden = (const float*)d_in[0];
  const int*   pos    = (const int*)d_in[1];
  const float* ln1w   = (const float*)d_in[2];
  const float* ln1b   = (const float*)d_in[3];
  const float* ln2w   = (const float*)d_in[4];
  const float* ln2b   = (const float*)d_in[5];
  const float* Wqkv   = (const float*)d_in[6];
  const float* Wout   = (const float*)d_in[7];
  const float* Wrouter= (const float*)d_in[8];
  const float* w1     = (const float*)d_in[9];
  const float* v1     = (const float*)d_in[10];
  const float* w2     = (const float*)d_in[11];
  float* out = (float*)d_out;

  char* w = (char*)d_ws;
  size_t o = 0;
  auto take = [&](size_t bytes) {
    size_t cur = o;
    o += (bytes + 255) & ~(size_t)255;
    return cur;
  };
  bf16*  wqkvT = (bf16*)(w + take((size_t)NQKV_ * D_ * 2));
  bf16*  woutT = (bf16*)(w + take((size_t)D_ * D_ * 2));
  bf16*  w1t   = (bf16*)(w + take((size_t)E_ * D_ * F_ * 2));
  bf16*  v1t   = (bf16*)(w + take((size_t)E_ * D_ * F_ * 2));
  bf16*  w2t   = (bf16*)(w + take((size_t)E_ * F_ * D_ * 2));
  bf16*  h1ln  = (bf16*)(w + take((size_t)S_ * D_ * 2));
  float* qkv   = (float*)(w + take((size_t)S_ * NQKV_ * 4));
  bf16*  qb    = (bf16*)(w + take((size_t)S_ * H_ * HD_ * 2));
  bf16*  kb    = (bf16*)(w + take((size_t)S_ * HKV_ * HD_ * 2));
  bf16*  vb    = (bf16*)(w + take((size_t)S_ * HKV_ * HD_ * 2));
  bf16*  attnb = (bf16*)(w + take((size_t)S_ * D_ * 2));
  float* aproj = (float*)(w + take((size_t)S_ * D_ * 4));
  float* h2    = (float*)(w + take((size_t)S_ * D_ * 4));
  bf16*  h2ln  = (bf16*)(w + take((size_t)S_ * D_ * 2));
  int*   tokE  = (int*)(w + take((size_t)S_ * 2 * 4));
  float* tokW  = (float*)(w + take((size_t)S_ * 2 * 4));
  int*   counts= (int*)(w + take(8 * 4));
  int*   offs  = (int*)(w + take(9 * 4));
  int*   cursor= (int*)(w + take(8 * 4));
  int*   perm  = (int*)(w + take((size_t)2 * S_ * 4));
  float* pairw = (float*)(w + take((size_t)2 * S_ * 4));
  int*   ppos  = (int*)(w + take((size_t)2 * S_ * 4));
  bf16*  inter = (bf16*)(w + take((size_t)2 * S_ * F_ * 2));
  float* pout  = (float*)(w + take((size_t)2 * S_ * D_ * 4));
  (void)ws_size; (void)n_in; (void)in_sizes; (void)out_size;

  // 1. weight convert + transpose to bf16 [N][K]
  {
    int n = D_ * NQKV_;
    k_cvtT<<<(n + 255) / 256, 256, 0, stream>>>(Wqkv, wqkvT, D_, NQKV_);
    n = D_ * D_;
    k_cvtT<<<(n + 255) / 256, 256, 0, stream>>>(Wout, woutT, D_, D_);
    for (int e = 0; e < E_; ++e) {
      n = D_ * F_;
      k_cvtT<<<(n + 255) / 256, 256, 0, stream>>>(w1 + (size_t)e * D_ * F_,
                                                  w1t + (size_t)e * F_ * D_, D_, F_);
      k_cvtT<<<(n + 255) / 256, 256, 0, stream>>>(v1 + (size_t)e * D_ * F_,
                                                  v1t + (size_t)e * F_ * D_, D_, F_);
      k_cvtT<<<(n + 255) / 256, 256, 0, stream>>>(w2 + (size_t)e * F_ * D_,
                                                  w2t + (size_t)e * D_ * F_, F_, D_);
    }
  }
  // 2. LN1
  k_addln<<<S_, 256, 0, stream>>>(hidden, nullptr, nullptr, ln1w, ln1b, h1ln);
  // 3. QKV GEMM with clip
  k_gemm<<<dim3(S_ / 128, NQKV_ / 128), 256, 0, stream>>>(h1ln, wqkvT, qkv,
                                                          S_, NQKV_, D_, 8.0f);
  // 4. RoPE + split
  k_rope<<<dim3(S_, H_ + 2 * HKV_), 64, 0, stream>>>(qkv, pos, qb, kb, vb);
  // 5. flash attention
  k_fattn<<<dim3(S_ / 128, H_), 256, 0, stream>>>(qb, kb, vb, attnb);
  // 6. output projection
  k_gemm<<<dim3(S_ / 128, D_ / 128), 256, 0, stream>>>(attnb, woutT, aproj,
                                                       S_, D_, D_, 0.0f);
  // 7. residual + LN2
  k_addln<<<S_, 256, 0, stream>>>(hidden, aproj, h2, ln2w, ln2b, h2ln);
  // 8. router top-2
  k_router<<<S_, 256, 0, stream>>>(h2ln, Wrouter, tokE, tokW);
  // 9. build gathered pair lists
  hipMemsetAsync(counts, 0, 8 * sizeof(int), stream);
  k_count<<<S_ / 256, 256, 0, stream>>>(tokE, counts);
  k_offsets<<<1, 1, 0, stream>>>(counts, offs, cursor);
  k_scatter<<<S_ / 256, 256, 0, stream>>>(tokE, tokW, cursor, perm, pairw, ppos);
  // 10. grouped MoE GEMMs (worst-case grid; tiles beyond count exit early)
  k_moe_up<<<dim3(2 * S_ / 128, F_ / 128, E_), 256, 0, stream>>>(h2ln, w1t, v1t,
                                                                 perm, offs, inter);
  k_moe_down<<<dim3(2 * S_ / 128, D_ / 128, E_), 256, 0, stream>>>(inter, w2t,
                                                                   pairw, offs, pout);
  // 11. combine with residual
  k_combine<<<S_, 256, 0, stream>>>(h2, pout, ppos, out);
}
